// MultiHeadAttentionLayer_53420803227767
// MI455X (gfx1250) — matrix operand order repaired and checked by reading, not compiled, not run on hardware
//
#include <hip/hip_runtime.h>
#include <hip/hip_bf16.h>
#include <stddef.h>

// ---------------------------------------------------------------------------
// MultiHeadAttentionLayer for MI455X (gfx1250), wave32 + bf16 WMMA.
// B=32, S=1024, H=512.
// Pipeline: cvt(x,W) -> proj (WMMA) -> attention (WMMA+softmax) -> normalize
// ---------------------------------------------------------------------------

#define BATCH 32
#define SLEN  1024
#define HDIM  512

typedef __attribute__((ext_vector_type(16))) __bf16 v16bf;
typedef __attribute__((ext_vector_type(8)))  __bf16 v8bf;
typedef __attribute__((ext_vector_type(8)))  float  v8f;

// fp32 -> bf16 (RNE), returning raw bits
static __device__ __forceinline__ unsigned short f2bf_u(float f) {
    union { float f; unsigned u; } v; v.f = f;
    return (unsigned short)((v.u + 0x7FFFu + ((v.u >> 16) & 1u)) >> 16);
}
static __device__ __forceinline__ __bf16 f2bf(float f) {
    unsigned short s = f2bf_u(f);
    __bf16 b;
    __builtin_memcpy(&b, &s, 2);
    return b;
}

// A-operand loader (contiguous K order, matching B):
// lane holds M = lane&15; hs = lane>>4 selects K = hs*16..+15.
static __device__ __forceinline__ v16bf load_a_row(const __bf16* rowp, int k0, int hs) {
    const v8bf lo = *(const v8bf*)(rowp + k0 + hs * 16);
    const v8bf hi = *(const v8bf*)(rowp + k0 + hs * 16 + 8);
    return __builtin_shufflevector(lo, hi, 0,1,2,3,4,5,6,7,8,9,10,11,12,13,14,15);
}

static __device__ __forceinline__ v8f wmma_bf16(v16bf a, v16bf b, v8f c) {
    return __builtin_amdgcn_wmma_f32_16x16x32_bf16(
        false, a, false, b, (short)0, c, false, false);
}

// ---------------------------------------------------------------------------
// Kernel 0: zero the per-batch denominator accumulator
// ---------------------------------------------------------------------------
__global__ void zero_denom_kernel(float* __restrict__ denom) {
    denom[threadIdx.x] = 0.0f;
}

// ---------------------------------------------------------------------------
// Kernel 0b: bulk fp32 -> bf16 conversion, 4 elements/thread (b128 in, b64 out)
// ---------------------------------------------------------------------------
__global__ __launch_bounds__(256) void cvt_kernel(
    const float* __restrict__ src, __bf16* __restrict__ dst, int n4)
{
    int i = blockIdx.x * 256 + threadIdx.x;
    if (i < n4) {
        float4 f = ((const float4*)src)[i];
        uint2 p;
        p.x = (unsigned)f2bf_u(f.x) | ((unsigned)f2bf_u(f.y) << 16);
        p.y = (unsigned)f2bf_u(f.z) | ((unsigned)f2bf_u(f.w) << 16);
        ((uint2*)dst)[i] = p;
    }
}

// ---------------------------------------------------------------------------
// Kernel 1: projections  Q/K = x @ W^T + b  (row-major bf16),
//                        V   -> stored TRANSPOSED: Vt[b][o][s] bf16
// Q is pre-scaled by 1/sqrt(H) (equivalent to scaling the energies).
// grid.x = (B*S)/64 row blocks, grid.y = 3 (q,k,v). block = 256 (8 waves).
// ---------------------------------------------------------------------------
__global__ __launch_bounds__(256) void proj_kernel(
    const __bf16* __restrict__ xb,
    const __bf16* __restrict__ Wqb, const float* __restrict__ bq,
    const __bf16* __restrict__ Wkb, const float* __restrict__ bk,
    const __bf16* __restrict__ Wvb, const float* __restrict__ bv,
    __bf16* __restrict__ Qb, __bf16* __restrict__ Kb, __bf16* __restrict__ Vt)
{
    __shared__ __bf16 Xs[64][HDIM + 8];   // 64 rows of x (bf16), 65KB

    const int proj = blockIdx.y;
    const __bf16* W    = (proj == 0) ? Wqb : (proj == 1) ? Wkb : Wvb;
    const float*  bias = (proj == 0) ? bq  : (proj == 1) ? bk  : bv;
    const float   osc  = (proj == 0) ? 0.04419417382415922f : 1.0f; // 1/sqrt(512)

    const int row0 = blockIdx.x * 64;
    const int tid  = threadIdx.x;

    // stage x block -> LDS, 16B chunks
    for (int i = tid; i < (64 * HDIM) / 8; i += 256) {
        int e = i * 8;
        int r = e >> 9, c = e & (HDIM - 1);
        *(uint4*)&Xs[r][c] = *(const uint4*)&xb[(size_t)(row0 + r) * HDIM + c];
    }
    __syncthreads();

    const int wave = tid >> 5, lane = tid & 31;
    const int lr = lane & 15, hs = lane >> 4;

    // 4 m-tiles x 32 n-tiles; each wave keeps a fixed m-tile (A rows reused)
    for (int t = wave; t < 128; t += 8) {
        const int mtile = t & 3;
        const int o0    = (t >> 2) * 16;
        v8f acc = {};
        const __bf16* arow = &Xs[mtile * 16 + lr][0];
        for (int k0 = 0; k0 < HDIM; k0 += 32) {
            v16bf a = load_a_row(arow, k0, hs);
            v16bf b = *(const v16bf*)&W[(size_t)(o0 + lr) * HDIM + k0 + hs * 16];
            acc = wmma_bf16(a, b, acc);
        }
        const float bo  = bias[o0 + lr];
        const int   gr0 = row0 + mtile * 16;
        if (proj < 2) {
            __bf16* dst = (proj == 0) ? Qb : Kb;
            #pragma unroll
            for (int j = 0; j < 8; ++j) {
                int m = j + hs * 8;
                dst[(size_t)(gr0 + m) * HDIM + o0 + lr] = f2bf((acc[j] + bo) * osc);
            }
        } else {
            const int b_ = gr0 >> 10;               // batch (64 | 1024)
            #pragma unroll
            for (int j = 0; j < 8; ++j) {
                int m = j + hs * 8;
                int s = (gr0 + m) & (SLEN - 1);
                Vt[((size_t)b_ * HDIM + o0 + lr) * SLEN + s] = f2bf(acc[j] + bo);
            }
        }
    }
}

// ---------------------------------------------------------------------------
// Kernel 2: attention for one (batch, 32-query block).
// phase1: E = Q K^T -> LDS fp32 [32][1024]   (Q pre-scaled by 1/sqrt(H))
// phase2: row softmax; Es overwritten with exp(e-m) (single exp pass);
//         P = exp*mask (bf16); denom += rowsum(A)
// phase3: out = (P @ V) / Z   (global denom applied by kernel 3)
// grid = (32 qblocks, 32 batches), block = 256 (8 waves). ~233KB LDS.
// ---------------------------------------------------------------------------
__global__ __launch_bounds__(256) void attn_kernel(
    const __bf16* __restrict__ Qb, const __bf16* __restrict__ Kb,
    const __bf16* __restrict__ Vt, const float* __restrict__ mask,
    float* __restrict__ out, float* __restrict__ denom)
{
    __shared__ float  Es[32][SLEN + 8];    // energy strip, padded (132KB)
    __shared__ __bf16 Ps[32][SLEN + 16];   // masked probs bf16   (66.5KB)
    __shared__ __bf16 Qs[32][HDIM + 8];    // Q block bf16        (33.3KB)
    __shared__ float  red[32][8];
    __shared__ float  zinv[32];

    const int qb = blockIdx.x, b = blockIdx.y;
    const int q0 = qb * 32;
    const int tid = threadIdx.x;

    const __bf16* Qbt = Qb + (size_t)b * SLEN * HDIM;
    const __bf16* Kbt = Kb + (size_t)b * SLEN * HDIM;
    const __bf16* Vbt = Vt + (size_t)b * HDIM * SLEN;

    for (int i = tid; i < (32 * HDIM) / 8; i += 256) {
        int e = i * 8;
        int r = e >> 9, c = e & (HDIM - 1);
        *(uint4*)&Qs[r][c] = *(const uint4*)&Qbt[(size_t)(q0 + r) * HDIM + c];
    }

    // warm L2/L0 with this block's mask strip while WMMA phase runs
    {
        const int r = tid >> 3, l8 = tid & 7;
        const char* mp = (const char*)(mask + ((size_t)b * SLEN + q0 + r) * SLEN + l8 * 128);
        #pragma unroll
        for (int pb = 0; pb < 512; pb += 128)
            __builtin_prefetch(mp + pb, 0, 1);
    }
    __syncthreads();

    const int wave = tid >> 5, lane = tid & 31;
    const int lr = lane & 15, hs = lane >> 4;

    // ---- phase 1: E = Q K^T ------------------------------------------------
    for (int t = wave; t < 128; t += 8) {       // 2 m-tiles x 64 n-tiles
        const int mtile = t & 1;
        const int n0    = (t >> 1) * 16;
        v8f acc = {};
        const __bf16* arow = &Qs[mtile * 16 + lr][0];
        for (int k0 = 0; k0 < HDIM; k0 += 32) {
            v16bf a  = load_a_row(arow, k0, hs);
            v16bf bm = *(const v16bf*)&Kbt[(size_t)(n0 + lr) * HDIM + k0 + hs * 16];
            acc = wmma_bf16(a, bm, acc);
        }
        #pragma unroll
        for (int j = 0; j < 8; ++j)
            Es[mtile * 16 + j + hs * 8][n0 + lr] = acc[j];
    }
    __syncthreads();

    // ---- phase 2: softmax (single exp pass) + masked probs -----------------
    {
        const int r = tid >> 3, l8 = tid & 7;   // 8 threads per row
        float m = -1e30f;
        for (int i = 0; i < 128; ++i) m = fmaxf(m, Es[r][l8 * 128 + i]);
        red[r][l8] = m;  __syncthreads();
        if (l8 == 0) {
            float mm = red[r][0];
            #pragma unroll
            for (int i = 1; i < 8; ++i) mm = fmaxf(mm, red[r][i]);
            red[r][0] = mm;
        }
        __syncthreads();
        m = red[r][0];
        __syncthreads();

        float z = 0.0f;
        for (int i = 0; i < 128; ++i) {         // exp once, store back in place
            int c = l8 * 128 + i;
            float e = __expf(Es[r][c] - m);
            Es[r][c] = e;
            z += e;
        }
        red[r][l8] = z;  __syncthreads();
        if (l8 == 0) {
            float zz = 0.0f;
            #pragma unroll
            for (int i = 0; i < 8; ++i) zz += red[r][i];
            zinv[r] = 1.0f / zz;
        }
        __syncthreads();

        const float* mrow = mask + ((size_t)b * SLEN + q0 + r) * SLEN;
        float w = 0.0f;
        for (int i = 0; i < 128; i += 2) {      // packed bf16 pair stores
            int c = l8 * 128 + i;
            float pm0 = Es[r][c]     * mrow[c];
            float pm1 = Es[r][c + 1] * mrow[c + 1];
            unsigned pk = (unsigned)f2bf_u(pm0) | ((unsigned)f2bf_u(pm1) << 16);
            *(unsigned*)&Ps[r][c] = pk;
            w += pm0 + pm1;
        }
        red[r][l8] = w;  __syncthreads();
        if (l8 == 0) {
            float ww = 0.0f;
            #pragma unroll
            for (int i = 0; i < 8; ++i) ww += red[r][i];
            atomicAdd(&denom[b], ww * zinv[r]);  // sum over A for this batch
        }
    }
    __syncthreads();

    // ---- phase 3: out = (P @ V) * (1/Z) ------------------------------------
    for (int t = wave; t < 64; t += 8) {        // 2 m-tiles x 32 o-tiles
        const int mtile = t & 1;
        const int o0    = (t >> 1) * 16;
        v8f acc = {};
        const __bf16* arow = &Ps[mtile * 16 + lr][0];
        for (int k0 = 0; k0 < SLEN; k0 += 32) {
            v16bf a  = load_a_row(arow, k0, hs);
            v16bf bm = *(const v16bf*)&Vbt[(size_t)(o0 + lr) * SLEN + k0 + hs * 16];
            acc = wmma_bf16(a, bm, acc);
        }
        #pragma unroll
        for (int j = 0; j < 8; ++j) {
            int m = mtile * 16 + j + hs * 8;
            out[((size_t)b * SLEN + q0 + m) * HDIM + o0 + lr] = acc[j] * zinv[m];
        }
    }
}

// ---------------------------------------------------------------------------
// Kernel 3: divide by global per-batch denominator
// ---------------------------------------------------------------------------
__global__ __launch_bounds__(256) void norm_kernel(
    float* __restrict__ out, const float* __restrict__ denom, size_t n)
{
    size_t i = (size_t)blockIdx.x * 256 + threadIdx.x;
    if (i < n) {
        int b = (int)(i >> 19);                 // / (1024*512)
        out[i] *= (1.0f / denom[b]);
    }
}

// ---------------------------------------------------------------------------
extern "C" void kernel_launch(void* const* d_in, const int* in_sizes, int n_in,
                              void* d_out, int out_size, void* d_ws, size_t ws_size,
                              hipStream_t stream)
{
    (void)in_sizes; (void)n_in; (void)out_size; (void)ws_size;
    const float* x    = (const float*)d_in[0];
    const float* mask = (const float*)d_in[1];
    const float* Wq   = (const float*)d_in[2];
    const float* bq   = (const float*)d_in[3];
    const float* Wk   = (const float*)d_in[4];
    const float* bk   = (const float*)d_in[5];
    const float* Wv   = (const float*)d_in[6];
    const float* bv   = (const float*)d_in[7];
    float* out = (float*)d_out;

    // workspace layout (bytes)
    char* ws = (char*)d_ws;
    const size_t elems = (size_t)BATCH * SLEN * HDIM;        // 16.78M
    const size_t wsz   = (size_t)HDIM * HDIM;                // 262144
    __bf16* Qb    = (__bf16*)(ws);                           // 32 MB
    __bf16* Kb    = (__bf16*)(ws + elems * 2);               // 32 MB
    __bf16* Vt    = (__bf16*)(ws + elems * 4);               // 32 MB (transposed)
    __bf16* xbf   = (__bf16*)(ws + elems * 6);               // 32 MB
    __bf16* Wqb   = (__bf16*)(ws + elems * 8);               // 512 KB
    __bf16* Wkb   = (__bf16*)(ws + elems * 8 + wsz * 2);     // 512 KB
    __bf16* Wvb   = (__bf16*)(ws + elems * 8 + wsz * 4);     // 512 KB
    float*  denom = (float*) (ws + elems * 8 + wsz * 6);     // 128 B

    zero_denom_kernel<<<1, BATCH, 0, stream>>>(denom);

    // bulk fp32 -> bf16 for x and the three weight matrices
    cvt_kernel<<<(unsigned)(elems / 4 / 256), 256, 0, stream>>>(x,  xbf, (int)(elems / 4));
    cvt_kernel<<<(unsigned)(wsz   / 4 / 256), 256, 0, stream>>>(Wq, Wqb, (int)(wsz / 4));
    cvt_kernel<<<(unsigned)(wsz   / 4 / 256), 256, 0, stream>>>(Wk, Wkb, (int)(wsz / 4));
    cvt_kernel<<<(unsigned)(wsz   / 4 / 256), 256, 0, stream>>>(Wv, Wvb, (int)(wsz / 4));

    dim3 pgrid((BATCH * SLEN) / 64, 3);
    proj_kernel<<<pgrid, 256, 0, stream>>>(xbf, Wqb, bq, Wkb, bk, Wvb, bv, Qb, Kb, Vt);

    dim3 agrid(SLEN / 32, BATCH);
    attn_kernel<<<agrid, 256, 0, stream>>>(Qb, Kb, Vt, mask, out, denom);

    size_t n = elems;
    norm_kernel<<<(unsigned)((n + 255) / 256), 256, 0, stream>>>(out, denom, n);
}
